// MultiHeadAttention_49151605735504
// MI455X (gfx1250) — matrix operand order repaired and checked by reading, not compiled, not run on hardware
//
#include <hip/hip_runtime.h>
#include <hip/hip_bf16.h>

// ---------------------------------------------------------------------------
// Problem constants (from reference)
// ---------------------------------------------------------------------------
#define EMBED   2048
#define H_      32
#define HKV     8
#define DH      64
#define MLP_    8192
#define DCONV   4
#define QKV_DIM 3072            // DH*(H+2*HKV)
#define IN_OUT  11264           // QKV_DIM + MLP
#define OUT_IN  6144            // H*DH + MLP/2
#define G_      4               // H/HKV
#define B_      2
#define S_      2048
#define MROWS   4096            // B*S
#define SCALE_  0.125f          // 1/sqrt(64)

typedef __attribute__((ext_vector_type(16))) __bf16        bf16x16;
typedef __attribute__((ext_vector_type(8)))  float         f32x8;
typedef __attribute__((ext_vector_type(4)))  unsigned int  u32x4;
typedef __attribute__((ext_vector_type(8)))  int           i32x8;
typedef __attribute__((ext_vector_type(4)))  int           i32x4;

// Tensor Data Mover availability (device pass only; host pass parses fallback)
#if defined(__has_builtin)
#  if __has_builtin(__builtin_amdgcn_tensor_load_to_lds) && \
      __has_builtin(__builtin_amdgcn_s_wait_tensorcnt)
#    define HAVE_TDM 1
#  endif
#endif
#ifndef HAVE_TDM
#  define HAVE_TDM 0
#endif

__device__ __forceinline__ f32x8 wmma_bf16(bf16x16 a, bf16x16 b, f32x8 c) {
    return __builtin_amdgcn_wmma_f32_16x16x32_bf16(false, a, false, b,
                                                   (short)0, c, false, false);
}

// Build a 16-element bf16 fragment from two 16-byte chunks.
__device__ __forceinline__ bf16x16 ld2(const __bf16* p0, const __bf16* p1) {
    union { uint4 u[2]; bf16x16 v; } f;
    f.u[0] = *reinterpret_cast<const uint4*>(p0);
    f.u[1] = *reinterpret_cast<const uint4*>(p1);
    return f.v;
}

// ---------------------------------------------------------------------------
// fp32 -> bf16 conversion (grid-stride)
// ---------------------------------------------------------------------------
__global__ void cvt_f32_bf16(const float* __restrict__ in,
                             __bf16* __restrict__ out, long n) {
    long i = (long)blockIdx.x * blockDim.x + threadIdx.x;
    long stride = (long)gridDim.x * blockDim.x;
    for (; i < n; i += stride) out[i] = (__bf16)in[i];
}

// ---------------------------------------------------------------------------
// GEMM: C[M,N] = A[M,K](bf16) * B[N,K]^T(bf16) + bias[N]   (fp32 out)
// Block tile 128x128, K-step 32, 8 waves (2x4), each wave 64x32 (4x2 WMMA).
// Tiles staged global->LDS by the Tensor Data Mover (double buffered).
// ---------------------------------------------------------------------------
#define GTM 128
#define GTN 128
#define GTK 32
#define LPAD 40   // padded LDS row (bf16 elems): 80B = 20-bank stride, no conflicts

#if HAVE_TDM
// D# group 1: 2D tile 32(bf16) x 128 rows, row stride = K elems, LDS rows
// padded: after every 16 DWORDs (64B = 32 bf16) insert 4 DWORDs (16B = 8 bf16)
// -> LDS rows of LPAD=40 bf16, matching As/Bs.
__device__ __forceinline__ i32x8 tdm_desc_g1(int kstride) {
    i32x8 g1;
    g1[0] = (1 << 16)                       // data_size = 2 bytes
          | (1 << 20)                       // pad_enable
          | (3 << 22)                       // pad_interval: 16 DWORDs
          | (3 << 25);                      // pad_amount: 4 DWORDs
    g1[1] = (int)0xFFFF0000u;               // tensor_dim0 = 0x7fffffff (lo16<<16)
    g1[2] = (int)(0x00007FFFu | 0xFFFF0000u); // dim0 hi16 | tensor_dim1 lo16
    g1[3] = (int)(0x00007FFFu | (32u << 16)); // dim1 hi16 | tile_dim0 = 32
    g1[4] = 128;                            // tile_dim1 = 128, tile_dim2 = 0
    g1[5] = kstride;                        // tensor_dim0_stride (elems)
    g1[6] = 0;
    g1[7] = 0;
    return g1;
}

__device__ __forceinline__ void tdm_tile_load(const __bf16* gsrc, void* lds_dst,
                                              const i32x8& g1) {
    unsigned long long ga = (unsigned long long)(uintptr_t)gsrc;
    u32x4 g0;
    g0[0] = 1u;                                              // count=1, flags=0
    g0[1] = (unsigned int)(uintptr_t)lds_dst;                // LDS byte offset
    g0[2] = (unsigned int)ga;                                // global addr lo
    g0[3] = (unsigned int)((ga >> 32) & 0x01FFFFFFu) | 0x80000000u; // hi | type=2
    i32x4 gz = {0, 0, 0, 0};
#if __clang_major__ >= 23
    i32x8 gz8 = {0, 0, 0, 0, 0, 0, 0, 0};
    __builtin_amdgcn_tensor_load_to_lds(g0, g1, gz, gz, gz8, 0);
#else
    __builtin_amdgcn_tensor_load_to_lds(g0, g1, gz, gz, 0);
#endif
}
#endif // HAVE_TDM

__global__ __launch_bounds__(256)
void gemm_bf16_wmma(const __bf16* __restrict__ A, const __bf16* __restrict__ B,
                    const float* __restrict__ bias, float* __restrict__ C,
                    int M, int N, int K) {
    __shared__ __bf16 As[2][GTM][LPAD];
    __shared__ __bf16 Bs[2][GTN][LPAD];

    const int tid  = threadIdx.x;
    const int lane = tid & 31;
    const int wid  = tid >> 5;
    const int l15  = lane & 15;
    const int hsel = lane >> 4;          // 0: lanes 0-15, 1: lanes 16-31

    const int bm = blockIdx.y * GTM;
    const int bn = blockIdx.x * GTN;
    const int m_off = (wid >> 2) * 64;   // wave M offset within block tile
    const int n_off = (wid & 3) * 32;    // wave N offset within block tile

    f32x8 acc[4][2] = {};

    const int kbA = hsel ? 8 : 0;        // A-frag K base (elems)
    const int kbB = hsel ? 8 : 0;        // B-frag K base (elems)
    const int nsteps = K / GTK;

#if HAVE_TDM
    const i32x8 g1 = tdm_desc_g1(K);
    if (wid == 0) {                      // prologue: stage tile pair 0
        tdm_tile_load(A + (size_t)bm * K, &As[0][0][0], g1);
        tdm_tile_load(B + (size_t)bn * K, &Bs[0][0][0], g1);
    }
#endif

    for (int i = 0; i < nsteps; ++i) {
        const int buf = i & 1;
#if HAVE_TDM
        if (wid == 0) {
            if (i + 1 < nsteps) {        // stage next pair into other buffer
                const int k0 = (i + 1) * GTK;
                tdm_tile_load(A + (size_t)bm * K + k0, &As[buf ^ 1][0][0], g1);
                tdm_tile_load(B + (size_t)bn * K + k0, &Bs[buf ^ 1][0][0], g1);
                __builtin_amdgcn_s_wait_tensorcnt(2);  // pair i done (in-order)
            } else {
                __builtin_amdgcn_s_wait_tensorcnt(0);
            }
        }
        __syncthreads();                 // publish tile i to all waves
#else
        // synchronous fallback: 512 16B-chunks per 128x32 tile
        const int k0s = i * GTK;
#pragma unroll
        for (int t = 0; t < 2; ++t) {
            int c   = tid + t * 256;
            int row = c >> 2;
            int q   = (c & 3) * 8;
            *reinterpret_cast<uint4*>(&As[buf][row][q]) =
                *reinterpret_cast<const uint4*>(A + (size_t)(bm + row) * K + k0s + q);
            *reinterpret_cast<uint4*>(&Bs[buf][row][q]) =
                *reinterpret_cast<const uint4*>(B + (size_t)(bn + row) * K + k0s + q);
        }
        __syncthreads();
#endif

        // ---- fragment loads ----
        bf16x16 af[4], bfr[2];
#pragma unroll
        for (int mt = 0; mt < 4; ++mt) {
            int r = m_off + mt * 16 + l15;
            af[mt] = ld2(&As[buf][r][kbA], &As[buf][r][kbA + 16]);
        }
#pragma unroll
        for (int nt = 0; nt < 2; ++nt) {
            int r = n_off + nt * 16 + l15;
            bfr[nt] = ld2(&Bs[buf][r][kbB], &Bs[buf][r][kbB + 16]);
        }

        // ---- 8 WMMAs ----
#pragma unroll
        for (int mt = 0; mt < 4; ++mt)
#pragma unroll
            for (int nt = 0; nt < 2; ++nt)
                acc[mt][nt] = wmma_bf16(af[mt], bfr[nt], acc[mt][nt]);

        __syncthreads();                 // reads done before buffer reuse
    }

    // ---- epilogue: bias + fp32 store ----
#pragma unroll
    for (int mt = 0; mt < 4; ++mt)
#pragma unroll
        for (int nt = 0; nt < 2; ++nt) {
            int col  = bn + n_off + nt * 16 + l15;
            float bv = bias[col];
#pragma unroll
            for (int j = 0; j < 8; ++j) {
                int row = bm + m_off + mt * 16 + j + hsel * 8;
                C[(size_t)row * N + col] = acc[mt][nt][j] + bv;
            }
        }
}

// ---------------------------------------------------------------------------
// Depthwise causal conv over qkv channels + routing into bf16 q / k / v^T
//   qbf : [B, HKV, G, S, DH]   kbf : [B, HKV, S, DH]   vtbf : [B, HKV, DH, S]
// ---------------------------------------------------------------------------
__global__ void conv_route(const float* __restrict__ h,
                           const float* __restrict__ Wconv,
                           const float* __restrict__ bconv,
                           __bf16* __restrict__ qbf,
                           __bf16* __restrict__ kbf,
                           __bf16* __restrict__ vtbf) {
    int idx = blockIdx.x * blockDim.x + threadIdx.x;   // B*S*QKV_DIM threads
    int c = idx % QKV_DIM;
    int s = (idx / QKV_DIM) % S_;
    int b = idx / (QKV_DIM * S_);

    float acc = bconv[c];
    const float* col = h + (size_t)b * S_ * IN_OUT + c;
#pragma unroll
    for (int w = 0; w < DCONV; ++w) {
        int sp = s - (DCONV - 1) + w;
        if (sp >= 0) acc += col[(size_t)sp * IN_OUT] * Wconv[c * DCONV + w];
    }
    __bf16 v = (__bf16)acc;

    if (c < H_ * DH) {                       // Q: c = ((g*G + r)*DH + d)
        int g = c >> 8, r = (c >> 6) & 3, d = c & 63;
        qbf[((((size_t)(b * HKV + g)) * G_ + r) * S_ + s) * DH + d] = v;
    } else if (c < H_ * DH + HKV * DH) {     // K
        int g = (c - H_ * DH) >> 6, d = c & 63;
        kbf[(((size_t)(b * HKV + g)) * S_ + s) * DH + d] = v;
    } else {                                 // V (transposed)
        int g = (c - H_ * DH - HKV * DH) >> 6, d = c & 63;
        vtbf[(((size_t)(b * HKV + g)) * DH + d) * S_ + s] = v;
    }
}

// ---------------------------------------------------------------------------
// SwiGLU: cat[:, 2048 + j] = up * silu(gate)   (bf16 out, GEMM2 input)
// ---------------------------------------------------------------------------
__global__ void swiglu_kernel(const float* __restrict__ h,
                              __bf16* __restrict__ cat) {
    int idx = blockIdx.x * blockDim.x + threadIdx.x;   // B*S*(MLP/2)
    int j  = idx & 4095;
    int rs = idx >> 12;
    float up = h[(size_t)rs * IN_OUT + QKV_DIM + j];
    float g  = h[(size_t)rs * IN_OUT + QKV_DIM + MLP_ / 2 + j];
    float val = up * (g / (1.f + __expf(-g)));
    cat[(size_t)rs * OUT_IN + H_ * DH + j] = (__bf16)val;
}

// ---------------------------------------------------------------------------
// Causal GQA flash attention. One wave per (b, g, r, 16-query tile).
// Keys processed in blocks of 32; per block: 4 WMMA scores + 4 WMMA P*V.
// Per-wave LDS buffer transposes P from C-layout to A-layout.
// ---------------------------------------------------------------------------
__global__ __launch_bounds__(128)
void flash_attn(const __bf16* __restrict__ qbf, const __bf16* __restrict__ kbf,
                const __bf16* __restrict__ vtbf, __bf16* __restrict__ cat) {
    __shared__ __bf16 psh[4][16][LPAD];

    const int lane = threadIdx.x & 31;
    const int wid  = threadIdx.x >> 5;
    const int l15  = lane & 15;
    const int hsel = lane >> 4;

    const int w    = blockIdx.x * 4 + wid;   // 8192 waves total
    const int qt   = w & 127;                // query tile (16 rows)
    const int head = w >> 7;                 // ((b*HKV+g)*G + r), 0..63
    const int b    = head >> 5;
    const int gkv  = (head >> 2) & 7;

    const __bf16* qhead  = qbf  + (size_t)head * S_ * DH;
    const __bf16* khead  = kbf  + (size_t)(b * HKV + gkv) * S_ * DH;
    const __bf16* vthead = vtbf + (size_t)(b * HKV + gkv) * DH * S_;

    const int kbA = hsel ? 8 : 0;
    const int kbB = hsel ? 8 : 0;

    // Q fragments (16 rows x 64 d, two K=32 chunks)
    const __bf16* qp = qhead + (size_t)(qt * 16 + l15) * DH;
    bf16x16 qf0 = ld2(qp + kbA,      qp + kbA + 16);
    bf16x16 qf1 = ld2(qp + 32 + kbA, qp + 48 + kbA);

    f32x8 acc[4] = {};
    float mrow[8], lrow[8];
#pragma unroll
    for (int j = 0; j < 8; ++j) { mrow[j] = -3.0e38f; lrow[j] = 0.f; }

    const int ktmax = (qt * 16 + 15) >> 5;
    for (int kt = 0; kt <= ktmax; ++kt) {
        const int key0 = kt * 32;

        // prefetch next key block (K rows + V^T rows) into cache
        if (kt < ktmax) {
            __builtin_prefetch(khead + (size_t)(key0 + 32 + lane) * DH, 0, 1);
            __builtin_prefetch(vthead + (size_t)(lane << 1) * S_ + key0 + 32, 0, 1);
        }

        // ---- scores: S[16x32] = Q * K^T ----
        f32x8 s0 = {}, s1 = {};
        {
            const __bf16* kp0 = khead + (size_t)(key0 + l15) * DH;
            s0 = wmma_bf16(qf0, ld2(kp0 + kbB,      kp0 + kbB + 16),      s0);
            s0 = wmma_bf16(qf1, ld2(kp0 + 32 + kbB, kp0 + 32 + kbB + 16), s0);
            const __bf16* kp1 = khead + (size_t)(key0 + 16 + l15) * DH;
            s1 = wmma_bf16(qf0, ld2(kp1 + kbB,      kp1 + kbB + 16),      s1);
            s1 = wmma_bf16(qf1, ld2(kp1 + 32 + kbB, kp1 + 32 + kbB + 16), s1);
        }

        // ---- scale + causal mask (C-layout: row j+8*hsel, col l15 / l15+16) ----
#pragma unroll
        for (int j = 0; j < 8; ++j) {
            int qr = qt * 16 + j + hsel * 8;
            int c0 = key0 + l15;
            float v0 = s0[j] * SCALE_; if (c0 > qr)      v0 = -3.0e38f;
            float v1 = s1[j] * SCALE_; if (c0 + 16 > qr) v1 = -3.0e38f;
            s0[j] = v0; s1[j] = v1;
        }

        // ---- online softmax ----
        float mnew[8], alpha[8];
#pragma unroll
        for (int j = 0; j < 8; ++j) {
            float mx = fmaxf(s0[j], s1[j]);
            mx = fmaxf(mx, __shfl_xor(mx, 1, 16));
            mx = fmaxf(mx, __shfl_xor(mx, 2, 16));
            mx = fmaxf(mx, __shfl_xor(mx, 4, 16));
            mx = fmaxf(mx, __shfl_xor(mx, 8, 16));
            mnew[j]  = fmaxf(mrow[j], mx);
            alpha[j] = __expf(mrow[j] - mnew[j]);
            mrow[j]  = mnew[j];
        }
#pragma unroll
        for (int j = 0; j < 8; ++j) {
            float p0 = __expf(s0[j] - mnew[j]);
            float p1 = __expf(s1[j] - mnew[j]);
            float rs = p0 + p1;
            rs += __shfl_xor(rs, 1, 16);
            rs += __shfl_xor(rs, 2, 16);
            rs += __shfl_xor(rs, 4, 16);
            rs += __shfl_xor(rs, 8, 16);
            lrow[j] = lrow[j] * alpha[j] + rs;
            int pr = j + hsel * 8;
            psh[wid][pr][l15]      = (__bf16)p0;
            psh[wid][pr][l15 + 16] = (__bf16)p1;
#pragma unroll
            for (int dt = 0; dt < 4; ++dt) acc[dt][j] *= alpha[j];
        }
        asm volatile("s_wait_dscnt 0" ::: "memory");

        // ---- P (A-frag via LDS transpose) x V (B-frags from v^T) ----
        bf16x16 pf = ld2(&psh[wid][l15][kbA], &psh[wid][l15][kbA + 16]);
#pragma unroll
        for (int dt = 0; dt < 4; ++dt) {
            const __bf16* vp = vthead + (size_t)(dt * 16 + l15) * S_ + key0 + kbB;
            acc[dt] = wmma_bf16(pf, ld2(vp, vp + 16), acc[dt]);
        }
    }

    // ---- normalize and write ctx into cat (bf16) ----
#pragma unroll
    for (int j = 0; j < 8; ++j) lrow[j] = 1.0f / lrow[j];
    const int colbase = (head & 31) << 6;    // (g*G+r)*DH
#pragma unroll
    for (int dt = 0; dt < 4; ++dt) {
        int col = colbase + dt * 16 + l15;
#pragma unroll
        for (int j = 0; j < 8; ++j) {
            int srow = qt * 16 + j + hsel * 8;
            cat[((size_t)b * S_ + srow) * OUT_IN + col] = (__bf16)(acc[dt][j] * lrow[j]);
        }
    }
}

// ---------------------------------------------------------------------------
// Host launcher
// ---------------------------------------------------------------------------
extern "C" void kernel_launch(void* const* d_in, const int* in_sizes, int n_in,
                              void* d_out, int out_size, void* d_ws, size_t ws_size,
                              hipStream_t stream) {
    const float* x     = (const float*)d_in[0];
    const float* Win   = (const float*)d_in[1];
    const float* bin_  = (const float*)d_in[2];
    const float* Wconv = (const float*)d_in[3];
    const float* bconv = (const float*)d_in[4];
    const float* Wout  = (const float*)d_in[5];
    const float* bout  = (const float*)d_in[6];
    float* out = (float*)d_out;

    char* ws = (char*)d_ws;
    size_t off = 0;
    auto alloc = [&](size_t bytes) -> void* {
        void* p = ws + off;
        off += (bytes + 255) & ~(size_t)255;
        return p;
    };

    __bf16* xbf    = (__bf16*)alloc((size_t)MROWS * EMBED * 2);
    __bf16* Winbf  = (__bf16*)alloc((size_t)IN_OUT * EMBED * 2);
    __bf16* Woutbf = (__bf16*)alloc((size_t)EMBED * OUT_IN * 2);
    float*  h      = (float*) alloc((size_t)MROWS * IN_OUT * 4);
    __bf16* qbf    = (__bf16*)alloc((size_t)B_ * H_ * S_ * DH * 2);
    __bf16* kbf    = (__bf16*)alloc((size_t)B_ * HKV * S_ * DH * 2);
    __bf16* vtbf   = (__bf16*)alloc((size_t)B_ * HKV * DH * S_ * 2);
    __bf16* catbf  = (__bf16*)alloc((size_t)MROWS * OUT_IN * 2);

    // 1) fp32 -> bf16 conversions
    {
        long n;
        n = (long)MROWS * EMBED;
        cvt_f32_bf16<<<(int)((n + 255) / 256), 256, 0, stream>>>(x, xbf, n);
        n = (long)IN_OUT * EMBED;
        cvt_f32_bf16<<<(int)((n + 255) / 256), 256, 0, stream>>>(Win, Winbf, n);
        n = (long)EMBED * OUT_IN;
        cvt_f32_bf16<<<(int)((n + 255) / 256), 256, 0, stream>>>(Wout, Woutbf, n);
    }

    // 2) GEMM1: h[4096,11264] = xbf * Winbf^T + bin_
    {
        dim3 grid(IN_OUT / GTN, MROWS / GTM);
        gemm_bf16_wmma<<<grid, 256, 0, stream>>>(xbf, Winbf, bin_, h,
                                                 MROWS, IN_OUT, EMBED);
    }

    // 3) causal depthwise conv + q/k/v^T routing (bf16)
    {
        int n = B_ * S_ * QKV_DIM;
        conv_route<<<n / 256, 256, 0, stream>>>(h, Wconv, bconv, qbf, kbf, vtbf);
    }

    // 4) SwiGLU -> cat[:, 2048:6144]
    {
        int n = B_ * S_ * (MLP_ / 2);
        swiglu_kernel<<<n / 256, 256, 0, stream>>>(h, catbf);
    }

    // 5) flash attention -> cat[:, 0:2048]
    {
        int waves = B_ * H_ * (S_ / 16);        // 8192
        flash_attn<<<waves / 4, 128, 0, stream>>>(qbf, kbf, vtbf, catbf);
    }

    // 6) GEMM2: out[4096,2048] = catbf * Woutbf^T + bout
    {
        dim3 grid(EMBED / GTN, MROWS / GTM);
        gemm_bf16_wmma<<<grid, 256, 0, stream>>>(catbf, Woutbf, bout, out,
                                                 MROWS, EMBED, OUT_IN);
    }
}
